// Conv2d_int8_65249143161229
// MI455X (gfx1250) — 8-bit matrix operand order repaired and checked by reading, compiled, not run on hardware
//
#include <hip/hip_runtime.h>

typedef __attribute__((ext_vector_type(8))) int v8i;

#define QMAXF 127.0f
#define LDS_ROW 66
#define LDS_COLSTRIDE 16
#define LDS_ROWSTRIDE (LDS_ROW * LDS_COLSTRIDE)
#define LDS_BYTES (4 * LDS_ROWSTRIDE)


__global__ void init_bits_kernel(unsigned* bits) {
  if (threadIdx.x < 2) bits[threadIdx.x] = 0u;
}

__global__ void absmax_kernel(const float* __restrict__ p, int n, unsigned* out) {
  __shared__ unsigned red[256];
  unsigned m = 0u;
  for (int i = blockIdx.x * blockDim.x + threadIdx.x; i < n; i += gridDim.x * blockDim.x)
    m = max(m, __float_as_uint(fabsf(p[i])));
  red[threadIdx.x] = m;
  __syncthreads();
  for (int s = 128; s > 0; s >>= 1) {
    if ((int)threadIdx.x < s) red[threadIdx.x] = max(red[threadIdx.x], red[threadIdx.x + s]);
    __syncthreads();
  }
  if (threadIdx.x == 0) atomicMax(out, red[0]);
}

__global__ void quantw_kernel(const float* __restrict__ w,
                              const unsigned* __restrict__ bits,
                              int* __restrict__ qb) {
  int idx = blockIdx.x * 256 + threadIdx.x;
  float maxw = __uint_as_float(bits[1]);
  float invw = QMAXF / maxw;
  int v    = idx & 7;
  int lane = (idx >> 3) & 31;
  int tt   = (idx >> 8) % 3;
  int ot   = idx / 768;
  int o    = ot * 16 + (lane & 15);
  int hb   = lane >> 4;
  unsigned word = 0;
#pragma unroll
  for (int b = 0; b < 4; ++b) {
    int Kp = tt * 64 + ((v >> 1) << 4) + ((v & 1) << 2) + hb * 8 + b;
    int q = 0;
    if (Kp < 144) {
      int kh = Kp / 48;
      int r48 = Kp - kh * 48;
      int kw = r48 >> 4;
      int c  = r48 & 15;
      float qf = fminf(fmaxf(rintf(w[o * 144 + c * 9 + kh * 3 + kw] * invw), -QMAXF), QMAXF);
      q = (int)qf;
    }
    word |= ((unsigned)(q & 0xFF)) << (8 * b);
  }
  qb[idx] = (int)word;
}

__global__ void __launch_bounds__(256) conv_wmma_kernel(
    const float* __restrict__ x, const unsigned* __restrict__ bits,
    const int* __restrict__ qb, const float* __restrict__ bias,
    float* __restrict__ out) {
  __shared__ __align__(16) signed char lds2[LDS_BYTES];

  const int rp = blockIdx.x;
  const int n  = blockIdx.y;
  const int tid = threadIdx.x;
  const float maxx = __uint_as_float(bits[0]);
  const float maxw = __uint_as_float(bits[1]);
  const float invx = QMAXF / maxx;
  const float sxsw = (maxx / QMAXF) * (maxw / QMAXF);
  const int ho0 = rp * 2;

  for (int idx = tid; idx < 16 * 4 * LDS_ROW; idx += 256) {
    int c   = idx / (4 * LDS_ROW);
    int rem = idx - c * (4 * LDS_ROW);
    int r   = rem / LDS_ROW;
    int col = rem - r * LDS_ROW;
    int grow = ho0 - 1 + r;
    int gcol = col - 1;
    signed char q = 0;
    if (grow >= 0 && grow < 64 && gcol >= 0 && gcol < 64) {
      float qf = fminf(fmaxf(rintf(x[((n * 16 + c) * 64 + grow) * 64 + gcol] * invx),
                             -QMAXF), QMAXF);
      q = (signed char)(int)qf;
    }
    lds2[r * LDS_ROWSTRIDE + col * LDS_COLSTRIDE + c] = q;
  }
  __syncthreads();

  const int wv   = tid >> 5;
  const int lane = tid & 31;
  const int rsel = wv >> 2;
  const int wo0  = (wv & 3) * 16;
  const int m    = lane & 15;
  const int h    = lane >> 4;

  v8i bfrag[2][3];
#pragma unroll
  for (int ot = 0; ot < 2; ++ot)
#pragma unroll
    for (int t = 0; t < 3; ++t)
      bfrag[ot][t] = *(const v8i*)(qb + ((((ot * 3 + t) * 32) + lane) << 3));

  v8i acc0 = {};
  v8i acc1 = {};
#pragma unroll
  for (int t = 0; t < 3; ++t) {
    v8i a;
#pragma unroll
    for (int v = 0; v < 8; ++v) {
      const int kb = ((v >> 1) << 4) + ((v & 1) << 2) + (h << 3);
      const int K0 = t * 64 + kb;
      int word = 0;
      if (K0 < 144) {
        const int kh  = K0 / 48;
        const int r48 = K0 - kh * 48;
        const int kw  = r48 >> 4;
        const int c0  = r48 & 15;
        word = *(const int*)(lds2 + (rsel + kh) * LDS_ROWSTRIDE
                                  + (wo0 + m + kw) * LDS_COLSTRIDE + c0);
      }
      a[v] = word;
    }
    acc0 = __builtin_amdgcn_wmma_i32_16x16x64_iu8(true, a, true, bfrag[0][t], acc0, false, false);
    acc1 = __builtin_amdgcn_wmma_i32_16x16x64_iu8(true, a, true, bfrag[1][t], acc1, false, false);
  }

  const int row = ho0 + rsel;
  const float b0 = bias[m];
  const float b1 = bias[16 + m];
#pragma unroll
  for (int r = 0; r < 8; ++r) {
    const int wo = wo0 + r + h * 8;
    out[((n * 32 + m) * 64 + row) * 64 + wo]      = (float)acc0[r] * sxsw + b0;
    out[((n * 32 + 16 + m) * 64 + row) * 64 + wo] = (float)acc1[r] * sxsw + b1;
  }
}

extern "C" void kernel_launch(void* const* d_in, const int* in_sizes, int n_in,
                              void* d_out, int out_size, void* d_ws, size_t ws_size,
                              hipStream_t stream) {
  const float* x    = (const float*)d_in[0];
  const float* w    = (const float*)d_in[1];
  const float* bias = (const float*)d_in[2];
  float* out = (float*)d_out;

  char* ws = (char*)d_ws;
  unsigned* bits = (unsigned*)ws;
  int* qb = (int*)(ws + 64);

  init_bits_kernel<<<1, 64, 0, stream>>>(bits);
  absmax_kernel<<<64, 256, 0, stream>>>(x, 4 * 16 * 64 * 64, bits + 0);
  absmax_kernel<<<2, 256, 0, stream>>>(w, 32 * 16 * 9, bits + 1);
  quantw_kernel<<<6, 256, 0, stream>>>(w, bits, qb);
  conv_wmma_kernel<<<dim3(32, 4), 256, 0, stream>>>(x, bits, qb, bias, out);
}
